// HAN_70377334113131
// MI455X (gfx1250) — matrix operand order repaired and checked by reading, not compiled, not run on hardware
//
#include <hip/hip_runtime.h>
#include <hip/hip_bf16.h>

// HAN forward for MI455X (gfx1250, wave32, WMMA).
// Only the movie-destination path contributes to the output, so edge types
// (movie->director) and (movie->actor) are skipped entirely.

typedef __attribute__((ext_vector_type(16))) _Float16 v16h;
typedef __attribute__((ext_vector_type(8)))  float    v8f;

#define NM 200000
#define ND 50000
#define NA 150000
#define EN 500000

// ---------------- helpers ----------------
__device__ __forceinline__ unsigned enc_ord(float f) {
  unsigned u = __float_as_uint(f);
  return (u & 0x80000000u) ? ~u : (u | 0x80000000u);   // monotone float->uint
}
__device__ __forceinline__ float dec_ord(unsigned k) {
  unsigned u = (k & 0x80000000u) ? (k & 0x7FFFFFFFu) : ~k;
  return __uint_as_float(u);
}

__global__ void fill_f32_k(float* __restrict__ p, float v, int n) {
  int i = blockIdx.x * blockDim.x + threadIdx.x;
  if (i < n) p[i] = v;
}
__global__ void fill_u32_k(unsigned* __restrict__ p, unsigned v, int n) {
  int i = blockIdx.x * blockDim.x + threadIdx.x;
  if (i < n) p[i] = v;
}

// ---------------- projection: H = X[N,256] @ W[256,128] + b, via WMMA ----------------
// Block = 256 threads = 8 waves; each wave computes a 16-row x 128-col strip.
// W is converted to f16 and packed in LDS in B-fragment order:
//   B 32x16 f16, lane L (col = L%16, half = L/16), element j -> K = (j<8?0:16) + 8*half + (j&7) (within K-tile).
// A 16x32 f16, lane L (row = L%16, half = L/16), element j -> K = (j<8?0:16) + 8*half + (j&7).
__global__ __launch_bounds__(256) void proj_wmma_k(
    const float* __restrict__ X, const float* __restrict__ W,
    const float* __restrict__ bias, float* __restrict__ H, int Nrows)
{
  __shared__ _Float16 wfrag[8 * 8 * 32 * 16];  // [kt][nt][lane][16] = 64 KB
  const int tid = threadIdx.x;
  for (int ent = tid; ent < 8 * 8 * 32; ent += 256) {
    int kt = ent >> 8;
    int rem = ent & 255;
    int nt = rem >> 5;
    int ln = rem & 31;
    int half = ln >> 4, l = ln & 15;
    int col = nt * 16 + l;
    _Float16* dst = &wfrag[ent * 16];
#pragma unroll
    for (int j = 0; j < 16; ++j) {
      int K = kt * 32 + 8 * half + (j & 7) + (j >> 3) * 16;
      dst[j] = (_Float16)W[K * 128 + col];
    }
  }
  __syncthreads();

  const int wave = tid >> 5, lane = tid & 31;
  const int l = lane & 15, half = lane >> 4;
  const int rowbase = blockIdx.x * 128 + wave * 16;
  if (rowbase >= Nrows) return;  // wave-uniform: EXEC stays all-ones for WMMA

  v8f acc[8] = {};
  const int row = rowbase + l;
  const float* xrow = X + (size_t)row * 256;
  for (int kt = 0; kt < 8; ++kt) {
    const float* xb = xrow + kt * 32 + 8 * half;
    float4 p0 = *(const float4*)(xb);
    float4 p1 = *(const float4*)(xb + 4);
    float4 p2 = *(const float4*)(xb + 16);
    float4 p3 = *(const float4*)(xb + 20);
    v16h a;
    a[0] = (_Float16)p0.x;  a[1] = (_Float16)p0.y;  a[2] = (_Float16)p0.z;  a[3] = (_Float16)p0.w;
    a[4] = (_Float16)p1.x;  a[5] = (_Float16)p1.y;  a[6] = (_Float16)p1.z;  a[7] = (_Float16)p1.w;
    a[8] = (_Float16)p2.x;  a[9] = (_Float16)p2.y;  a[10] = (_Float16)p2.z; a[11] = (_Float16)p2.w;
    a[12] = (_Float16)p3.x; a[13] = (_Float16)p3.y; a[14] = (_Float16)p3.z; a[15] = (_Float16)p3.w;
#pragma unroll
    for (int nt = 0; nt < 8; ++nt) {
      v16h b = *(const v16h*)&wfrag[((kt * 8 + nt) * 32 + lane) * 16];
      acc[nt] = __builtin_amdgcn_wmma_f32_16x16x32_f16(
          false, a, false, b, (short)0, acc[nt], false, false);
    }
  }
  // C/D layout: VGPR r holds (M = r + 8*half, N = l) within 16x16 tile nt
#pragma unroll
  for (int nt = 0; nt < 8; ++nt) {
    int n = nt * 16 + l;
    float bv = bias[n];
#pragma unroll
    for (int r = 0; r < 8; ++r) {
      int m = r + 8 * half;
      H[(size_t)(rowbase + m) * 128 + n] = acc[nt][r] + bv;
    }
  }
}

// ---------------- per-(node, head) attention dot:  out[n,h] = sum_k H[n,h,k]*att[h,k] ----------------
__global__ void head_dot_k(const float* __restrict__ H, const float* __restrict__ att,
                           float* __restrict__ out, int Nn)
{
  int t = blockIdx.x * blockDim.x + threadIdx.x;
  if (t >= Nn * 8) return;
  int n = t >> 3, hh = t & 7;
  const float* hp = H + (size_t)n * 128 + hh * 16;
  const float* ap = att + hh * 16;
  float s = 0.f;
#pragma unroll
  for (int k = 0; k < 16; ++k) s += hp[k] * ap[k];
  out[t] = s;
}

// ---------------- edge pass A: logits + segment max ----------------
__global__ void edge_logit_max_k(const int* __restrict__ ei, const float* __restrict__ a_s,
                                 const float* __restrict__ a_d, float* __restrict__ logits,
                                 unsigned* __restrict__ mkey)
{
  int t = blockIdx.x * blockDim.x + threadIdx.x;
  if (t >= EN * 8) return;
  int e = t >> 3, hh = t & 7;
  int row = ei[e], col = ei[EN + e];
  float x = a_s[row * 8 + hh] + a_d[col * 8 + hh];
  x = (x > 0.f) ? x : 0.2f * x;  // leaky_relu(0.2)
  logits[t] = x;
  atomicMax(&mkey[col * 8 + hh], enc_ord(x));
}

// ---------------- edge pass B: exp + segment sum ----------------
__global__ void edge_expsum_k(const int* __restrict__ ei, float* __restrict__ logits,
                              const unsigned* __restrict__ mkey, float* __restrict__ ssum)
{
  int t = blockIdx.x * blockDim.x + threadIdx.x;
  if (t >= EN * 8) return;
  int e = t >> 3, hh = t & 7;
  int col = ei[EN + e];
  float m = dec_ord(mkey[col * 8 + hh]);
  float ex = expf(logits[t] - m);
  logits[t] = ex;
  atomicAdd(&ssum[col * 8 + hh], ex);
}

// ---------------- edge pass C: weighted message scatter ----------------
__global__ void edge_scatter_k(const int* __restrict__ ei, const float* __restrict__ logits,
                               const float* __restrict__ ssum, const float* __restrict__ Hsrc,
                               float* __restrict__ agg)
{
  int t = blockIdx.x * blockDim.x + threadIdx.x;
  if (t >= EN * 128) return;
  int e = t >> 7, c = t & 127, hh = c >> 4;
  int row = ei[e], col = ei[EN + e];
  float w = logits[e * 8 + hh] / (ssum[col * 8 + hh] + 1e-16f);
  atomicAdd(&agg[(size_t)col * 128 + c], Hsrc[(size_t)row * 128 + c] * w);
}

// ---------------- semantic: colsum[n] += sum_rows tanh( (relu(AGG) @ kW + kb)[row,n] ) ----------------
__global__ __launch_bounds__(256) void tanh_colsum_k(
    const float* __restrict__ AGG, const float* __restrict__ KW,
    const float* __restrict__ kb, float* __restrict__ colsum, int Nrows)
{
  __shared__ _Float16 wfrag[4 * 8 * 32 * 16];  // 32 KB
  const int tid = threadIdx.x;
  for (int ent = tid; ent < 4 * 8 * 32; ent += 256) {
    int kt = ent >> 8;
    int rem = ent & 255;
    int nt = rem >> 5;
    int ln = rem & 31;
    int half = ln >> 4, l = ln & 15;
    int col = nt * 16 + l;
    _Float16* dst = &wfrag[ent * 16];
#pragma unroll
    for (int j = 0; j < 16; ++j) {
      int K = kt * 32 + 8 * half + (j & 7) + (j >> 3) * 16;
      dst[j] = (_Float16)KW[K * 128 + col];
    }
  }
  __syncthreads();

  const int wave = tid >> 5, lane = tid & 31;
  const int l = lane & 15, half = lane >> 4;
  const int ntiles = Nrows >> 4;
  float kbv[8];
#pragma unroll
  for (int nt = 0; nt < 8; ++nt) kbv[nt] = kb[nt * 16 + l];

  float sums[8] = {0.f, 0.f, 0.f, 0.f, 0.f, 0.f, 0.f, 0.f};
  for (int tile = blockIdx.x * 8 + wave; tile < ntiles; tile += gridDim.x * 8) {
    v8f acc[8] = {};
    const int row = tile * 16 + l;
    const float* arow = AGG + (size_t)row * 128;
    for (int kt = 0; kt < 4; ++kt) {
      const float* xb = arow + kt * 32 + 8 * half;
      float4 p0 = *(const float4*)(xb);
      float4 p1 = *(const float4*)(xb + 4);
      float4 p2 = *(const float4*)(xb + 16);
      float4 p3 = *(const float4*)(xb + 20);
      v16h a;
      a[0]  = (_Float16)fmaxf(p0.x, 0.f); a[1]  = (_Float16)fmaxf(p0.y, 0.f);
      a[2]  = (_Float16)fmaxf(p0.z, 0.f); a[3]  = (_Float16)fmaxf(p0.w, 0.f);
      a[4]  = (_Float16)fmaxf(p1.x, 0.f); a[5]  = (_Float16)fmaxf(p1.y, 0.f);
      a[6]  = (_Float16)fmaxf(p1.z, 0.f); a[7]  = (_Float16)fmaxf(p1.w, 0.f);
      a[8]  = (_Float16)fmaxf(p2.x, 0.f); a[9]  = (_Float16)fmaxf(p2.y, 0.f);
      a[10] = (_Float16)fmaxf(p2.z, 0.f); a[11] = (_Float16)fmaxf(p2.w, 0.f);
      a[12] = (_Float16)fmaxf(p3.x, 0.f); a[13] = (_Float16)fmaxf(p3.y, 0.f);
      a[14] = (_Float16)fmaxf(p3.z, 0.f); a[15] = (_Float16)fmaxf(p3.w, 0.f);
#pragma unroll
      for (int nt = 0; nt < 8; ++nt) {
        v16h b = *(const v16h*)&wfrag[((kt * 8 + nt) * 32 + lane) * 16];
        acc[nt] = __builtin_amdgcn_wmma_f32_16x16x32_f16(
            false, a, false, b, (short)0, acc[nt], false, false);
      }
    }
#pragma unroll
    for (int nt = 0; nt < 8; ++nt) {
#pragma unroll
      for (int r = 0; r < 8; ++r) sums[nt] += tanhf(acc[nt][r] + kbv[nt]);
    }
  }
#pragma unroll
  for (int nt = 0; nt < 8; ++nt) atomicAdd(&colsum[nt * 16 + l], sums[nt]);
}

// ---------------- semantic softmax over the 2 metapaths ----------------
__global__ void semantic_score_k(const float* __restrict__ colsum0, const float* __restrict__ colsum1,
                                 const float* __restrict__ q, float Ninv, float* __restrict__ attn)
{
  __shared__ float r0[128], r1[128];
  int t = threadIdx.x;
  r0[t] = q[t] * colsum0[t] * Ninv;
  r1[t] = q[t] * colsum1[t] * Ninv;
  __syncthreads();
  for (int s = 64; s > 0; s >>= 1) {
    if (t < s) { r0[t] += r0[t + s]; r1[t] += r1[t + s]; }
    __syncthreads();
  }
  if (t == 0) {
    float s0 = r0[0], s1 = r1[0];
    float mm = fmaxf(s0, s1);
    float e0 = expf(s0 - mm), e1 = expf(s1 - mm);
    float inv = 1.f / (e0 + e1);
    attn[0] = e0 * inv;
    attn[1] = e1 * inv;
  }
}

// ---------------- final: out = (attn0*relu(agg0)+attn1*relu(agg1)) @ linW + linb ----------------
__global__ void final_out_k(const float* __restrict__ agg0, const float* __restrict__ agg1,
                            const float* __restrict__ attn, const float* __restrict__ linW,
                            const float* __restrict__ linb, float* __restrict__ out)
{
  int t = blockIdx.x * blockDim.x + threadIdx.x;
  if (t >= NM * 16) return;
  int n = t >> 4, o = t & 15;
  float a0 = attn[0], a1 = attn[1];
  float acc = linb[o];
  const float* r0 = agg0 + (size_t)n * 128;
  const float* r1 = agg1 + (size_t)n * 128;
#pragma unroll 4
  for (int j = 0; j < 128; ++j) {
    float f = a0 * fmaxf(r0[j], 0.f) + a1 * fmaxf(r1[j], 0.f);
    acc += f * linW[j * 16 + o];
  }
  out[t] = acc;
}

// ---------------- launch ----------------
extern "C" void kernel_launch(void* const* d_in, const int* in_sizes, int n_in,
                              void* d_out, int out_size, void* d_ws, size_t ws_size,
                              hipStream_t stream)
{
  const float* x_movie    = (const float*)d_in[0];
  const float* x_director = (const float*)d_in[1];
  const float* x_actor    = (const float*)d_in[2];
  // d_in[3] = ei_movie_director (unused), d_in[5] = ei_movie_actor (unused)
  const int*   ei_dm      = (const int*)d_in[4];   // director -> movie
  const int*   ei_am      = (const int*)d_in[6];   // actor -> movie
  const float* W_movie    = (const float*)d_in[7];
  const float* b_movie    = (const float*)d_in[8];
  const float* W_director = (const float*)d_in[9];
  const float* b_director = (const float*)d_in[10];
  const float* W_actor    = (const float*)d_in[11];
  const float* b_actor    = (const float*)d_in[12];
  const float* att_src_dm = (const float*)d_in[15];
  const float* att_dst_dm = (const float*)d_in[16];
  const float* att_src_am = (const float*)d_in[19];
  const float* att_dst_am = (const float*)d_in[20];
  const float* q          = (const float*)d_in[21];
  const float* kW         = (const float*)d_in[22];
  const float* kb         = (const float*)d_in[23];
  const float* linW       = (const float*)d_in[24];
  const float* linb       = (const float*)d_in[25];
  float* out = (float*)d_out;

  size_t off = 0;
  auto alloc = [&](size_t bytes) -> void* {
    off = (off + 255) & ~(size_t)255;
    void* p = (char*)d_ws + off;
    off += bytes;
    return p;
  };
  float*    h_movie    = (float*)alloc((size_t)NM * 128 * 4);
  float*    h_director = (float*)alloc((size_t)ND * 128 * 4);
  float*    h_actor    = (float*)alloc((size_t)NA * 128 * 4);
  float*    agg0       = (float*)alloc((size_t)NM * 128 * 4);  // director->movie
  float*    agg1       = (float*)alloc((size_t)NM * 128 * 4);  // actor->movie
  float*    logits0    = (float*)alloc((size_t)EN * 8 * 4);
  float*    logits1    = (float*)alloc((size_t)EN * 8 * 4);
  float*    a_src_dm   = (float*)alloc((size_t)ND * 8 * 4);
  float*    a_dst_dm   = (float*)alloc((size_t)NM * 8 * 4);
  float*    a_src_am   = (float*)alloc((size_t)NA * 8 * 4);
  float*    a_dst_am   = (float*)alloc((size_t)NM * 8 * 4);
  float*    s0         = (float*)alloc((size_t)NM * 8 * 4);
  float*    s1         = (float*)alloc((size_t)NM * 8 * 4);
  unsigned* mkey0      = (unsigned*)alloc((size_t)NM * 8 * 4);
  unsigned* mkey1      = (unsigned*)alloc((size_t)NM * 8 * 4);
  float*    colsum0    = (float*)alloc(128 * 4);
  float*    colsum1    = (float*)alloc(128 * 4);
  float*    attn       = (float*)alloc(2 * 4);

  const int BLK = 256;
  auto grid = [](long long n) { return dim3((unsigned)((n + 255) / 256)); };

  // init accumulators (ws is poisoned by harness; we re-init every call)
  fill_f32_k<<<grid((long long)NM * 128), BLK, 0, stream>>>(agg0, 0.f, NM * 128);
  fill_f32_k<<<grid((long long)NM * 128), BLK, 0, stream>>>(agg1, 0.f, NM * 128);
  fill_f32_k<<<grid(NM * 8), BLK, 0, stream>>>(s0, 0.f, NM * 8);
  fill_f32_k<<<grid(NM * 8), BLK, 0, stream>>>(s1, 0.f, NM * 8);
  fill_u32_k<<<grid(NM * 8), BLK, 0, stream>>>(mkey0, 0u, NM * 8);
  fill_u32_k<<<grid(NM * 8), BLK, 0, stream>>>(mkey1, 0u, NM * 8);
  fill_f32_k<<<grid(128), BLK, 0, stream>>>(colsum0, 0.f, 128);
  fill_f32_k<<<grid(128), BLK, 0, stream>>>(colsum1, 0.f, 128);

  // projections (WMMA)
  proj_wmma_k<<<dim3((NM + 127) / 128), BLK, 0, stream>>>(x_movie, W_movie, b_movie, h_movie, NM);
  proj_wmma_k<<<dim3((ND + 127) / 128), BLK, 0, stream>>>(x_director, W_director, b_director, h_director, ND);
  proj_wmma_k<<<dim3((NA + 127) / 128), BLK, 0, stream>>>(x_actor, W_actor, b_actor, h_actor, NA);

  // per-head attention dots
  head_dot_k<<<grid(ND * 8), BLK, 0, stream>>>(h_director, att_src_dm, a_src_dm, ND);
  head_dot_k<<<grid(NM * 8), BLK, 0, stream>>>(h_movie, att_dst_dm, a_dst_dm, NM);
  head_dot_k<<<grid(NA * 8), BLK, 0, stream>>>(h_actor, att_src_am, a_src_am, NA);
  head_dot_k<<<grid(NM * 8), BLK, 0, stream>>>(h_movie, att_dst_am, a_dst_am, NM);

  // edge softmax + scatter, director->movie
  edge_logit_max_k<<<grid((long long)EN * 8), BLK, 0, stream>>>(ei_dm, a_src_dm, a_dst_dm, logits0, mkey0);
  edge_expsum_k<<<grid((long long)EN * 8), BLK, 0, stream>>>(ei_dm, logits0, mkey0, s0);
  edge_scatter_k<<<grid((long long)EN * 128), BLK, 0, stream>>>(ei_dm, logits0, s0, h_director, agg0);

  // edge softmax + scatter, actor->movie
  edge_logit_max_k<<<grid((long long)EN * 8), BLK, 0, stream>>>(ei_am, a_src_am, a_dst_am, logits1, mkey1);
  edge_expsum_k<<<grid((long long)EN * 8), BLK, 0, stream>>>(ei_am, logits1, mkey1, s1);
  edge_scatter_k<<<grid((long long)EN * 128), BLK, 0, stream>>>(ei_am, logits1, s1, h_actor, agg1);

  // semantic attention (WMMA GEMM + tanh + column reduction)
  tanh_colsum_k<<<dim3(256), BLK, 0, stream>>>(agg0, kW, kb, colsum0, NM);
  tanh_colsum_k<<<dim3(256), BLK, 0, stream>>>(agg1, kW, kb, colsum1, NM);
  semantic_score_k<<<dim3(1), dim3(128), 0, stream>>>(colsum0, colsum1, q, 1.0f / (float)NM, attn);

  // fuse + final linear
  final_out_k<<<grid((long long)NM * 16), BLK, 0, stream>>>(agg0, agg1, attn, linW, linb, out);

  (void)in_sizes; (void)n_in; (void)out_size; (void)ws_size;
}
